// PKM_43310450213618
// MI455X (gfx1250) — compile-verified, matrix operand order checked by reading, not run on hardware
//
#include <hip/hip_runtime.h>
#include <stdint.h>

// ---------------------------------------------------------------------------
// PKM attention for MI455X (gfx1250, wave32, WMMA, async global->LDS).
// fp32 GEMMs run as split-bf16 (hi+lo) emulation on v_wmma_f32_16x16x32_bf16:
//   a*b ~= a_hi*b_hi + a_hi*b_lo + a_lo*b_hi   (3 WMMAs per K=32 block)
// The hi/lo split is materialized ONCE per matrix as separate bf16 planes in
// global memory (hi+lo = 4B/elem, same bytes as f32), so GEMM inner loops are
// pure:  global_load_async_to_lds_b128 -> ds_load_b128 -> v_wmma  with no
// conversion VALU on the critical path. K-slabs are double-buffered in LDS
// with ASYNCcnt/s_wait_asynccnt overlap.
// ---------------------------------------------------------------------------

typedef __attribute__((ext_vector_type(16))) __bf16 v16bf;
typedef __attribute__((ext_vector_type(8)))  float  v8f;
typedef __attribute__((ext_vector_type(4)))  float  v4f;
typedef __attribute__((ext_vector_type(4)))  unsigned int v4u;
typedef __attribute__((ext_vector_type(2)))  unsigned int v2u;
typedef unsigned short u16;

#define DIM      1024
#define HEADS    8
#define NUM_KEYS 256
#define TOPK     32
#define D_HEAD   128
#define BATCH    2
#define SEQ      2048
#define BT       (BATCH * SEQ)   // 4096 token rows

#define BSTR     40   // LDS bf16 row stride: 32 data + 8 pad (80B, 16B aligned)
#define DSTRIDE  264  // LDS dots row stride in f32: 256 data + 8 pad

// --------------------- async global->LDS (CDNA5, ASYNCcnt) ------------------
// ISA 15.18.3: GLOBAL_LOAD_ASYNC_TO_LDS_B128  vdst(LDS byte offset VGPR),
// vaddr(64b global address), no SADDR ("off"). Completion via s_wait_asynccnt.
__device__ __forceinline__ unsigned lds_off(const void* p) {
  return (unsigned)(uintptr_t)p;   // flat LDS aperture: low 32 bits are the LDS offset
}
__device__ __forceinline__ void async_b128(unsigned lds_byte_off, const void* g) {
  asm volatile("global_load_async_to_lds_b128 %0, %1, off"
               :: "v"(lds_byte_off), "v"((unsigned long long)(uintptr_t)g)
               : "memory");
}
#define ASYNC_WAIT(n) asm volatile("s_wait_asynccnt " #n ::: "memory")

// ----------------------------- bf16 split helpers ---------------------------
__device__ __forceinline__ u16 f32_to_bf16_rne(float f) {
  unsigned u = __float_as_uint(f);
  unsigned r = u + 0x7FFFu + ((u >> 16) & 1u);   // round-to-nearest-even
  return (u16)(r >> 16);
}
__device__ __forceinline__ float bf16_bits_to_f32(u16 s) {
  return __uint_as_float(((unsigned)s) << 16);
}
__device__ __forceinline__ void split2(float f, u16& h, u16& l) {
  h = f32_to_bf16_rne(f);
  l = f32_to_bf16_rne(f - bf16_bits_to_f32(h));
}

// WMMA fragment load: 16 contiguous bf16 per lane (two ds_load_b128).
// A-frag 16x32: lanes 0-15 -> rows M, K=0..15; lanes 16-31 -> K=16..31.
// B-frag 32x16: lanes hold column n, 16 contiguous K values (same k order as A, not the ISA table order).
union FragU { v4f f[2]; v16bf v; };
__device__ __forceinline__ v16bf load_frag(const u16* p) {
  FragU u;
  u.f[0] = *(const v4f*)(p);
  u.f[1] = *(const v4f*)(p + 8);
  return u.v;
}

__device__ __forceinline__ v8f wmma3(v16bf ah, v16bf al, v16bf bh, v16bf bl, v8f c) {
  c = __builtin_amdgcn_wmma_f32_16x16x32_bf16(false, ah, false, bh, (short)0, c, false, false);
  c = __builtin_amdgcn_wmma_f32_16x16x32_bf16(false, ah, false, bl, (short)0, c, false, false);
  c = __builtin_amdgcn_wmma_f32_16x16x32_bf16(false, al, false, bh, (short)0, c, false, false);
  return c;
}

// ---------------------------------------------------------------------------
// Kernel 0: split an f32 array into bf16 hi/lo planes (8 elems per thread).
// ---------------------------------------------------------------------------
__global__ __launch_bounds__(256)
void split_planes(const float* __restrict__ src, u16* __restrict__ hi,
                  u16* __restrict__ lo, int n8) {
  const int idx = blockIdx.x * 256 + threadIdx.x;
  if (idx >= n8) return;
  const float* p = src + (size_t)idx * 8;
  unsigned hw[4], lw[4];
#pragma unroll
  for (int q = 0; q < 2; ++q) {
    v4f d = *(const v4f*)(p + 4 * q);
    u16 hs[4], ls[4];
#pragma unroll
    for (int e = 0; e < 4; ++e) split2(d[e], hs[e], ls[e]);
    hw[2 * q]     = (unsigned)hs[0] | ((unsigned)hs[1] << 16);
    hw[2 * q + 1] = (unsigned)hs[2] | ((unsigned)hs[3] << 16);
    lw[2 * q]     = (unsigned)ls[0] | ((unsigned)ls[1] << 16);
    lw[2 * q + 1] = (unsigned)ls[2] | ((unsigned)ls[3] << 16);
  }
  v4u H = {hw[0], hw[1], hw[2], hw[3]};
  v4u L = {lw[0], lw[1], lw[2], lw[3]};
  *(v4u*)(hi + (size_t)idx * 8) = H;
  *(v4u*)(lo + (size_t)idx * 8) = L;
}

// ---------------------------------------------------------------------------
// Kernels 1/4: C[M,N] = A[M,K] * B[N,K]^T from bf16 hi/lo planes.
// Block tile 128x128, 8 waves -> 32x64 each (2x4 tiles of 16x16).
// Double-buffered async K-slabs (32 wide): 4 planes x 128x32 bf16.
// OUT_MODE: 1 = f32 + bias, 2 = write hi/lo bf16 planes (feeds next GEMM).
// ---------------------------------------------------------------------------
template <int OUT_MODE>
__global__ __launch_bounds__(256)
void gemm_nt_bf16x3(const u16* __restrict__ Ah, const u16* __restrict__ Al,
                    const u16* __restrict__ Bh, const u16* __restrict__ Bl,
                    const float* __restrict__ bias, float* __restrict__ Cf,
                    u16* __restrict__ Ch, u16* __restrict__ Cl, int N, int K) {
  extern __shared__ u16 smem[];
  const int TILE = 128 * BSTR;
  u16* sAh = smem;                 // [2][TILE]
  u16* sAl = smem + 2 * TILE;
  u16* sBh = smem + 4 * TILE;
  u16* sBl = smem + 6 * TILE;

  const int tid  = threadIdx.x;
  const int lane = tid & 31;
  const int w    = tid >> 5;
  const int bm   = blockIdx.y * 128;
  const int bn   = blockIdx.x * 128;
  const int r0   = (w >> 1) * 32;
  const int c0   = (w & 1) * 64;
  const int lrow = lane & 15;
  const int lko  = (lane >> 4) * 16;

  const v8f vzero = {0, 0, 0, 0, 0, 0, 0, 0};
  v8f acc[2][4];
#pragma unroll
  for (int mi = 0; mi < 2; ++mi)
#pragma unroll
    for (int ni = 0; ni < 4; ++ni) acc[mi][ni] = vzero;

  // 8 async b128 per thread per slab: 4 planes x 128 rows x 32 bf16.
  auto issue_slab = [&](int k0, int buf) {
    u16* dAh = sAh + buf * TILE;
    u16* dAl = sAl + buf * TILE;
    u16* dBh = sBh + buf * TILE;
    u16* dBl = sBl + buf * TILE;
#pragma unroll
    for (int i = 0; i < 2; ++i) {
      int idx = tid + i * 256;           // 0..511 -> 128 rows x 4 b128-chunks
      int row = idx >> 2;
      int c8  = (idx & 3) * 8;
      size_t ga = (size_t)(bm + row) * K + k0 + c8;
      size_t gb = (size_t)(bn + row) * K + k0 + c8;
      unsigned lo = row * BSTR + c8;
      async_b128(lds_off(dAh + lo), Ah + ga);
      async_b128(lds_off(dAl + lo), Al + ga);
      async_b128(lds_off(dBh + lo), Bh + gb);
      async_b128(lds_off(dBl + lo), Bl + gb);
    }
  };

  issue_slab(0, 0);
  const int KB = K >> 5;
  for (int kb = 0; kb < KB; ++kb) {
    const int buf = kb & 1;
    if (kb + 1 < KB) {
      issue_slab((kb + 1) << 5, buf ^ 1);
      ASYNC_WAIT(8);                     // slab kb done (async loads retire in order)
    } else {
      ASYNC_WAIT(0);
    }
    __syncthreads();                     // all waves' slab-kb data visible in LDS

    const u16* tAh = sAh + buf * TILE;
    const u16* tAl = sAl + buf * TILE;
    const u16* tBh = sBh + buf * TILE;
    const u16* tBl = sBl + buf * TILE;

    v16bf ah[2], al[2];
#pragma unroll
    for (int mi = 0; mi < 2; ++mi) {
      const int rr = (r0 + mi * 16 + lrow) * BSTR + lko;
      ah[mi] = load_frag(tAh + rr);
      al[mi] = load_frag(tAl + rr);
    }
#pragma unroll
    for (int ni = 0; ni < 4; ++ni) {
      const int cc = (c0 + ni * 16 + lrow) * BSTR + lko;
      v16bf bh = load_frag(tBh + cc);
      v16bf bl = load_frag(tBl + cc);
#pragma unroll
      for (int mi = 0; mi < 2; ++mi)
        acc[mi][ni] = wmma3(ah[mi], al[mi], bh, bl, acc[mi][ni]);
    }
    __syncthreads();                     // buffer 'buf' overwritten next iter
  }

  // C/D layout: VGPR r -> M = r + 8*(lane>>4), N = lane&15 (ISA 7.12.2)
  const int orow = (lane >> 4) * 8;
  const int ocol = lane & 15;
#pragma unroll
  for (int mi = 0; mi < 2; ++mi)
#pragma unroll
    for (int ni = 0; ni < 4; ++ni) {
      const int col = bn + c0 + ni * 16 + ocol;
      const float bb = (OUT_MODE == 1) ? bias[col] : 0.0f;
#pragma unroll
      for (int r = 0; r < 8; ++r) {
        const int row = bm + r0 + mi * 16 + orow + r;
        const float f = acc[mi][ni][r];
        if (OUT_MODE == 1) {
          Cf[(size_t)row * N + col] = f + bb;
        } else {
          u16 h, l;
          split2(f, h, l);
          Ch[(size_t)row * N + col] = h;
          Cl[(size_t)row * N + col] = l;
        }
      }
    }
}

// ---------------------------------------------------------------------------
// Kernel 2: dots[b,h,t,p,n] = q[p,b,t,h,:] . keys[h,n,p,:]  (K=128, N=256)
// fused with top-32-of-256 per row. One block per (64 token rows, h, p).
// Operands are bf16 hi/lo planes; async double-buffered K-slabs.
// ---------------------------------------------------------------------------
__global__ __launch_bounds__(256)
void dots_topk(const u16* __restrict__ Qh, const u16* __restrict__ Ql,
               const u16* __restrict__ Kh, const u16* __restrict__ Kl,
               float* __restrict__ Sout, int* __restrict__ Iout) {
  extern __shared__ u16 smem[];
  const int ATILE = 64 * BSTR, BTILE = 256 * BSTR;
  u16* sAh = smem;                         // [2][ATILE]
  u16* sAl = smem + 2 * ATILE;
  u16* sBh = smem + 4 * ATILE;             // [2][BTILE]
  u16* sBl = smem + 4 * ATILE + 2 * BTILE;
  float* sDots = (float*)(smem + 4 * ATILE + 4 * BTILE);  // [64][DSTRIDE]

  const int tid   = threadIdx.x;
  const int lane  = tid & 31;
  const int w     = tid >> 5;
  const int mbase = blockIdx.x * 64;       // token-row base in [0,4096)
  const int hp    = blockIdx.y;
  const int h     = hp >> 1;
  const int p     = hp & 1;

  // q row m: plane[m, p*1024 + h*128 + k]  (row stride 2*DIM)
  const size_t aofs = (size_t)p * DIM + (size_t)h * D_HEAD;
  // keys row n: plane[((h*256+n)*2+p)*128 + k]  (row stride 2*D_HEAD)
  const size_t bofs = (size_t)h * NUM_KEYS * 2 * D_HEAD + (size_t)p * D_HEAD;

  const int r0   = (w >> 1) * 16;          // 4 bands of 16 rows
  const int c0   = (w & 1) * 128;          // 2 halves of 128 cols
  const int lrow = lane & 15;
  const int lko  = (lane >> 4) * 16;

  const v8f vzero = {0, 0, 0, 0, 0, 0, 0, 0};
  v8f acc[8];
#pragma unroll
  for (int ni = 0; ni < 8; ++ni) acc[ni] = vzero;

  // 10 async b128 per thread per slab: A 2 (2 planes) + B 8 (2 planes x 4).
  auto issue_slab = [&](int k0, int buf) {
    u16* dAh = sAh + buf * ATILE;
    u16* dAl = sAl + buf * ATILE;
    u16* dBh = sBh + buf * BTILE;
    u16* dBl = sBl + buf * BTILE;
    {
      int row = tid >> 2;                  // 0..63
      int c8  = (tid & 3) * 8;
      size_t ga = aofs + (size_t)(mbase + row) * (2 * DIM) + k0 + c8;
      unsigned lo = row * BSTR + c8;
      async_b128(lds_off(dAh + lo), Qh + ga);
      async_b128(lds_off(dAl + lo), Ql + ga);
    }
#pragma unroll
    for (int i = 0; i < 4; ++i) {
      int idx = tid + i * 256;             // 0..1023 -> 256 rows x 4 chunks
      int row = idx >> 2;
      int c8  = (idx & 3) * 8;
      size_t gb = bofs + (size_t)row * (2 * D_HEAD) + k0 + c8;
      unsigned lo = row * BSTR + c8;
      async_b128(lds_off(dBh + lo), Kh + gb);
      async_b128(lds_off(dBl + lo), Kl + gb);
    }
  };

  issue_slab(0, 0);
  const int KB = D_HEAD >> 5;  // 4
  for (int kb = 0; kb < KB; ++kb) {
    const int buf = kb & 1;
    if (kb + 1 < KB) {
      issue_slab((kb + 1) << 5, buf ^ 1);
      ASYNC_WAIT(10);
    } else {
      ASYNC_WAIT(0);
    }
    __syncthreads();

    const u16* tAh = sAh + buf * ATILE;
    const u16* tAl = sAl + buf * ATILE;
    const u16* tBh = sBh + buf * BTILE;
    const u16* tBl = sBl + buf * BTILE;

    const int rr = (r0 + lrow) * BSTR + lko;
    v16bf ah = load_frag(tAh + rr);
    v16bf al = load_frag(tAl + rr);
#pragma unroll
    for (int ni = 0; ni < 8; ++ni) {
      const int cc = (c0 + ni * 16 + lrow) * BSTR + lko;
      v16bf bh = load_frag(tBh + cc);
      v16bf bl = load_frag(tBl + cc);
      acc[ni] = wmma3(ah, al, bh, bl, acc[ni]);
    }
    __syncthreads();
  }

  const int orow = (lane >> 4) * 8;
  const int ocol = lane & 15;
#pragma unroll
  for (int ni = 0; ni < 8; ++ni)
#pragma unroll
    for (int r = 0; r < 8; ++r)
      sDots[(r0 + orow + r) * DSTRIDE + c0 + ni * 16 + ocol] = acc[ni][r];
  __syncthreads();

  // top-32-of-256 per row; wave w handles rows [w*8, w*8+8)
  for (int rr2 = 0; rr2 < 8; ++rr2) {
    const int row = w * 8 + rr2;
    float v[8];
#pragma unroll
    for (int j = 0; j < 8; ++j) v[j] = sDots[row * DSTRIDE + lane + 32 * j];
    float outS = 0.f;
    int outI = 0;
    for (int sel = 0; sel < TOPK; ++sel) {
      float m = v[0];
      int mj = 0;
#pragma unroll
      for (int j = 1; j < 8; ++j)
        if (v[j] > m) { m = v[j]; mj = j; }
      float bmv = m;
      int bi = (mj << 5) | lane;        // candidate index n = 32*j + lane
#pragma unroll
      for (int off = 16; off > 0; off >>= 1) {  // wave argmax (low-index ties)
        float ov = __shfl_xor(bmv, off, 32);
        int   oi = __shfl_xor(bi, off, 32);
        if (ov > bmv || (ov == bmv && oi < bi)) { bmv = ov; bi = oi; }
      }
      if (lane == sel) { outS = bmv; outI = bi; }
      if ((bi & 31) == lane) {
        const int jw = bi >> 5;
#pragma unroll
        for (int j = 0; j < 8; ++j)
          if (j == jw) v[j] = -3.0e38f;
      }
    }
    const int gm = mbase + row;         // = b*SEQ + t
    const int b  = gm >> 11;
    const int t  = gm & (SEQ - 1);
    const size_t outRow = (((size_t)b * HEADS + h) * SEQ + t) * 2 + p;
    Sout[outRow * TOPK + lane] = outS;
    Iout[outRow * TOPK + lane] = outI;
  }
}

// ---------------------------------------------------------------------------
// Kernel 3: combine s0+s1 (32x32 candidates), top-32, softmax, gather+reduce.
// One wave per (b,h,tt); output written as bf16 hi/lo planes for GEMM4.
// ---------------------------------------------------------------------------
__global__ __launch_bounds__(256)
void combine_topk_gather(const float* __restrict__ S, const int* __restrict__ I,
                         const float* __restrict__ values,
                         u16* __restrict__ Oh, u16* __restrict__ Ol) {
  const int tid  = threadIdx.x;
  const int lane = tid & 31;
  const int gw   = blockIdx.x * 8 + (tid >> 5);   // 0 .. BATCH*HEADS*SEQ-1
  const int b    = gw >> 14;                      // / (HEADS*SEQ)
  const int rem  = gw & 16383;
  const int h    = rem >> 11;
  const int tt   = rem & (SEQ - 1);
  const int i    = tt >> 1;
  const int p    = tt & 1;

  const size_t base = ((size_t)b * HEADS + h) * SEQ;
  const size_t row0 = (base + i) * 2 + p;               // token i (first half)
  const size_t row1 = (base + SEQ / 2 + i) * 2 + p;     // token 1024+i

  const float su = S[row0 * TOPK + lane];
  const int   iu = I[row0 * TOPK + lane];
  const float sv = S[row1 * TOPK + lane];
  const int   iv = I[row1 * TOPK + lane];

  unsigned mask = 0u;   // per-lane consumed v's
  float fs = 0.f;
  int fu = 0, fv = 0;
  for (int sel = 0; sel < TOPK; ++sel) {
    const int j = __builtin_ctz(~mask);           // best remaining v (sv sorted desc)
    float cand = su + __shfl(sv, j, 32);
    int ci = (lane << 5) | j;                     // flat index u*32+v
#pragma unroll
    for (int off = 16; off > 0; off >>= 1) {
      float ov = __shfl_xor(cand, off, 32);
      int   oi = __shfl_xor(ci, off, 32);
      if (ov > cand || (ov == cand && oi < ci)) { cand = ov; ci = oi; }
    }
    if (lane == sel) { fs = cand; fu = ci >> 5; fv = ci & 31; }
    if (lane == (ci >> 5)) mask |= (1u << (ci & 31));
  }

  const int vidx = __shfl(iu, fu, 32) * NUM_KEYS + __shfl(iv, fv, 32);

  // softmax over the 32 winners (fs at lane 0 is the max)
  const float mx = __shfl(fs, 0, 32);
  float e = __expf(fs - mx);
  float s = e;
#pragma unroll
  for (int off = 16; off > 0; off >>= 1) s += __shfl_xor(s, off, 32);
  const float attn = e / s;

  // Gather 32 x 128-float rows from the 256MB values table.
  const float* vbase = values + (size_t)h * NUM_KEYS * NUM_KEYS * D_HEAD;
#pragma unroll
  for (int sel = 0; sel < TOPK; ++sel) {          // prefetch all rows first
    const int vi = __shfl(vidx, sel, 32);
    __builtin_prefetch(vbase + (size_t)vi * D_HEAD + lane * 4, 0, 0);
  }
  v4f acc = {0, 0, 0, 0};
  for (int sel = 0; sel < TOPK; ++sel) {
    const float wgt = __shfl(attn, sel, 32);
    const int vi = __shfl(vidx, sel, 32);
    const v4f vv = *(const v4f*)(vbase + (size_t)vi * D_HEAD + lane * 4);
    acc += wgt * vv;
  }

  // inter layout (b, t, h*128+d), written as hi/lo bf16 planes.
  u16 hs[4], ls[4];
#pragma unroll
  for (int e2 = 0; e2 < 4; ++e2) split2(acc[e2], hs[e2], ls[e2]);
  v2u H = {(unsigned)hs[0] | ((unsigned)hs[1] << 16), (unsigned)hs[2] | ((unsigned)hs[3] << 16)};
  v2u L = {(unsigned)ls[0] | ((unsigned)ls[1] << 16), (unsigned)ls[2] | ((unsigned)ls[3] << 16)};
  const size_t off = ((size_t)b * SEQ + tt) * DIM + (size_t)h * D_HEAD + lane * 4;
  *(v2u*)(Oh + off) = H;
  *(v2u*)(Ol + off) = L;
}

// ---------------------------------------------------------------------------
extern "C" void kernel_launch(void* const* d_in, const int* in_sizes, int n_in,
                              void* d_out, int out_size, void* d_ws, size_t ws_size,
                              hipStream_t stream) {
  const float* x      = (const float*)d_in[0];  // (2,2048,1024)
  const float* Wq     = (const float*)d_in[1];  // (2048,1024)
  const float* keys   = (const float*)d_in[2];  // (8,256,2,128)
  const float* values = (const float*)d_in[3];  // (8,65536,128)
  const float* Wo     = (const float*)d_in[4];  // (1024,1024)
  const float* bo     = (const float*)d_in[5];  // (1024,)
  float* out = (float*)d_out;                   // (2,2048,1024)

  // ---- workspace layout (~94 MB): f32/i32 region first, bf16 planes after.
  const size_t nX   = (size_t)BT * DIM;          // 4,194,304
  const size_t nWq  = (size_t)2 * DIM * DIM;     // 2,097,152
  const size_t nKey = (size_t)HEADS * NUM_KEYS * 2 * D_HEAD;  // 524,288
  const size_t nWo  = (size_t)DIM * DIM;         // 1,048,576
  const size_t nQ   = (size_t)BT * 2 * DIM;      // 8,388,608
  const size_t nSc  = (size_t)BATCH * HEADS * SEQ * 2 * TOPK; // 2,097,152

  float* Sbuf = (float*)d_ws;
  int*   Ibuf = (int*)(Sbuf + nSc);
  u16* up = (u16*)(Ibuf + nSc);
  u16* xh   = up;             u16* xl   = xh + nX;
  u16* Wqh  = xl + nX;        u16* Wql  = Wqh + nWq;
  u16* keyh = Wql + nWq;      u16* keyl = keyh + nKey;
  u16* Woh  = keyl + nKey;    u16* Wol  = Woh + nWo;
  u16* qph  = Wol + nWo;      u16* qpl  = qph + nQ;
  u16* ih   = qpl + nQ;       u16* il   = ih + nX;

  const size_t smemG = (size_t)8 * 128 * BSTR * sizeof(u16);   // 81,920 B
  const size_t smem2 = ((size_t)4 * 64 * BSTR + 4 * 256 * BSTR) * sizeof(u16)
                     + (size_t)64 * DSTRIDE * sizeof(float);   // 169,984 B

  // 0) split inputs into bf16 hi/lo planes (streaming, ~3us total)
  split_planes<<<dim3((unsigned)(nX   / 8 / 256)), 256, 0, stream>>>(x,    xh,   xl,   (int)(nX / 8));
  split_planes<<<dim3((unsigned)(nWq  / 8 / 256)), 256, 0, stream>>>(Wq,   Wqh,  Wql,  (int)(nWq / 8));
  split_planes<<<dim3((unsigned)(nKey / 8 / 256)), 256, 0, stream>>>(keys, keyh, keyl, (int)(nKey / 8));
  split_planes<<<dim3((unsigned)(nWo  / 8 / 256)), 256, 0, stream>>>(Wo,   Woh,  Wol,  (int)(nWo / 8));

  // 1) q = x @ Wq^T  (4096 x 2048 x 1024) -> hi/lo planes for kernel 2
  gemm_nt_bf16x3<2><<<dim3((2 * DIM) / 128, BT / 128), 256, smemG, stream>>>(
      xh, xl, Wqh, Wql, nullptr, nullptr, qph, qpl, 2 * DIM, DIM);

  // 2) dots + first top-k
  dots_topk<<<dim3(BT / 64, HEADS * 2), 256, smem2, stream>>>(
      qph, qpl, keyh, keyl, Sbuf, Ibuf);

  // 3) combine + second top-k + softmax + gather -> hi/lo planes for GEMM4
  combine_topk_gather<<<dim3((BATCH * HEADS * SEQ) / 8), 256, 0, stream>>>(
      Sbuf, Ibuf, values, ih, il);

  // 4) out = inter @ Wo^T + bo   (4096 x 1024 x 1024)
  gemm_nt_bf16x3<1><<<dim3(DIM / 128, BT / 128), 256, smemG, stream>>>(
      ih, il, Woh, Wol, bo, out, nullptr, nullptr, DIM, DIM);

  (void)in_sizes; (void)n_in; (void)out_size; (void)ws_size;
}
